// ColorFlowBlock_47141561040939
// MI455X (gfx1250) — matrix operand order repaired and checked by reading, not compiled, not run on hardware
//
#include <hip/hip_runtime.h>

typedef __attribute__((ext_vector_type(16))) __bf16 bf16x16;
typedef __attribute__((ext_vector_type(8)))  __bf16 bf16x8;
typedef __attribute__((ext_vector_type(8)))  float  f32x8;

#define WMMA_BF16(a, b, c) \
  __builtin_amdgcn_wmma_f32_16x16x32_bf16(false, (a), false, (b), (short)0, (c), false, false)

#define EROWS 48   // edges per block   (LDS: 48*576*2 = 55296 B)
#define NROWS 48   // nodes per block   (LDS: 48*544*2 = 52224 B)
#define OROWS 64   // nodes per block in output projection (64*256*2 = 32768 B)

__device__ __forceinline__ float silu_f(float x) { return x / (1.0f + __expf(-x)); }

// A-operand fetch from row-major bf16 LDS row, per CDNA5 16-bit A 16x32 layout:
// lane group g = lane>>4: halves 0..7 -> K = g*8 + i ; halves 8..15 -> K = 16 + g*8 + (i-8)
__device__ __forceinline__ bf16x16 load_a_lds(const __bf16* p) {
  bf16x8 lo = *(const bf16x8*)p;         // K = g*8 .. g*8+7
  bf16x8 hi = *(const bf16x8*)(p + 16);  // K = 16+g*8 .. 16+g*8+7
  return __builtin_shufflevector(lo, hi, 0, 1, 2, 3, 4, 5, 6, 7,
                                 8, 9, 10, 11, 12, 13, 14, 15);
}

// ---------------------------------------------------------------------------
// Weight swizzle: fp32 [K, ncols] row-major  ->  bf16 WMMA-B tiles.
// dst element idx = ((kt*NT + nt)*32 + lane)*16 + i
// maps (k = kt*32 + (lane>>4)*8 + (i&7) + (i>>3)*16, n = nt*16 + (lane&15)); k >= Kact -> 0.
// ---------------------------------------------------------------------------
__global__ void swizzle_kernel(const float* __restrict__ src, __bf16* __restrict__ dst,
                               int Kact, int NT, int ncols, int total) {
  int idx = blockIdx.x * blockDim.x + threadIdx.x;
  if (idx >= total) return;
  const int i = idx & 15;
  const int lane = (idx >> 4) & 31;
  const int tile = idx >> 9;
  const int nt = tile % NT;
  const int kt = tile / NT;
  const int g = lane >> 4;
  const int n = nt * 16 + (lane & 15);
  const int k = kt * 32 + g * 8 + (i & 7) + ((i >> 3) << 4);
  float v = (k < Kact) ? src[(size_t)k * ncols + n] : 0.0f;
  dst[idx] = (__bf16)v;
}

__global__ void zero_kernel(float* __restrict__ p, size_t n) {
  size_t i = (size_t)blockIdx.x * blockDim.x + threadIdx.x;
  size_t stride = (size_t)gridDim.x * blockDim.x;
  for (; i < n; i += stride) p[i] = 0.0f;
}

// ---------------------------------------------------------------------------
// Input projection: h = concat(scalars16, col_emb8, role_emb8) @ in_w[32,256] + b
// ---------------------------------------------------------------------------
__global__ __launch_bounds__(256) void input_proj_kernel(
    const float* __restrict__ scalars, const int* __restrict__ node_col,
    const int* __restrict__ node_role, const float* __restrict__ cemb,
    const float* __restrict__ remb, const float* __restrict__ w,
    const float* __restrict__ b, float* __restrict__ h,
    __bf16* __restrict__ hbf, int N) {
  const int node = blockIdx.x;
  if (node >= N) return;
  const int t = threadIdx.x;
  __shared__ float nin[32];
  if (t < 16)      nin[t] = scalars[(size_t)node * 16 + t];
  else if (t < 24) nin[t] = cemb[node_col[node] * 8 + (t - 16)];
  else if (t < 32) nin[t] = remb[node_role[node] * 8 + (t - 24)];
  __syncthreads();
  float acc = b[t];
#pragma unroll
  for (int k = 0; k < 32; k++) acc += nin[k] * w[k * 256 + t];
  h[(size_t)node * 256 + t] = acc;
  hbf[(size_t)node * 256 + t] = (__bf16)acc;
}

// ---------------------------------------------------------------------------
// Edge MLP + fused atomic scatter. EROWS edges / block, 8 waves.
// edge_in (560 -> pad 576) staged bf16 in LDS; GEMM1 K=576 (18 kt), GEMM2 K=256.
// ---------------------------------------------------------------------------
__global__ __launch_bounds__(256) void edge_mlp_kernel(
    const __bf16* __restrict__ hbf, const int* __restrict__ edge_index,
    const int* __restrict__ edge_rel, const int* __restrict__ node_role,
    const int* __restrict__ node_col, const float* __restrict__ rel_e,
    const float* __restrict__ role_e, const float* __restrict__ col_e,
    const __bf16* __restrict__ w1, const float* __restrict__ b1,
    const __bf16* __restrict__ w2, const float* __restrict__ b2,
    float* __restrict__ agg, int E) {
  __shared__ __align__(16) __bf16 sA[EROWS * 576];  // reused as m1 (EROWS x 256)
  __shared__ int ssrc[EROWS], sdst[EROWS];
  const int tid = threadIdx.x;
  const int eb = blockIdx.x * EROWS;
  if (tid < EROWS) {
    int e = eb + tid; if (e >= E) e = E - 1;
    ssrc[tid] = edge_index[e];
    sdst[tid] = edge_index[E + e];
  }
  __syncthreads();
  // stage h[src] | h[dst] (bf16, uint4 copies, flat index)
  for (int idx = tid; idx < EROWS * 32; idx += 256) {
    const int e = idx >> 5, i = idx & 31;
    const uint4* hs = (const uint4*)(hbf + (size_t)ssrc[e] * 256);
    const uint4* hd = (const uint4*)(hbf + (size_t)sdst[e] * 256);
    uint4* row = (uint4*)(sA + e * 576);
    row[i] = hs[i];
    row[32 + i] = hd[i];
  }
  if (tid < EROWS) {  // tail: rel(16) role_s(8) role_d(8) col_s(8) col_d(8) pad(16)
    const int e = tid;
    int ge = eb + e; if (ge >= E) ge = E - 1;
    __bf16* t = sA + e * 576 + 512;
    const float* re = rel_e + edge_rel[ge] * 16;
#pragma unroll
    for (int i = 0; i < 16; i++) t[i] = (__bf16)re[i];
    const int s = ssrc[e], d = sdst[e];
    const float* r0 = role_e + node_role[s] * 8;
    const float* r1 = role_e + node_role[d] * 8;
    const float* c0 = col_e + node_col[s] * 8;
    const float* c1 = col_e + node_col[d] * 8;
#pragma unroll
    for (int i = 0; i < 8; i++) {
      t[16 + i] = (__bf16)r0[i]; t[24 + i] = (__bf16)r1[i];
      t[32 + i] = (__bf16)c0[i]; t[40 + i] = (__bf16)c1[i];
    }
#pragma unroll
    for (int i = 0; i < 16; i++) t[48 + i] = (__bf16)0.0f;
  }
  __syncthreads();

  const int lane = tid & 31, wv = tid >> 5;
  const int g = lane >> 4, ln = lane & 15;
  const int nt0 = wv * 2;  // 8 waves x 2 n-tiles = 16 tiles = 256 cols

  f32x8 acc[3][2] = {};
  for (int kt = 0; kt < 18; ++kt) {
    const __bf16* bp = w1 + ((size_t)((kt * 16 + nt0) * 32 + lane)) * 16;
    __builtin_prefetch(bp + 16 * 32 * 16, 0, 0);  // next kt's B tiles -> global_prefetch
    bf16x16 bA = *(const bf16x16*)bp;
    bf16x16 bB = *(const bf16x16*)(bp + 512);
#pragma unroll
    for (int mt = 0; mt < 3; mt++) {
      bf16x16 a = load_a_lds(sA + (mt * 16 + ln) * 576 + kt * 32 + g * 8);
      acc[mt][0] = WMMA_BF16(a, bA, acc[mt][0]);
      acc[mt][1] = WMMA_BF16(a, bB, acc[mt][1]);
    }
  }
  __syncthreads();  // all waves done reading edge_in before overwriting with m1
  __bf16* m1 = sA;
#pragma unroll
  for (int mt = 0; mt < 3; mt++)
#pragma unroll
    for (int nj = 0; nj < 2; nj++) {
      const int n = (nt0 + nj) * 16 + ln;
      const float bias = b1[n];
#pragma unroll
      for (int j = 0; j < 8; j++) {
        float x = acc[mt][nj][j] + bias;
        m1[(mt * 16 + g * 8 + j) * 256 + n] = (__bf16)silu_f(x);
      }
    }
  __syncthreads();

  f32x8 acc2[3][2] = {};
  for (int kt = 0; kt < 8; ++kt) {
    const __bf16* bp = w2 + ((size_t)((kt * 16 + nt0) * 32 + lane)) * 16;
    bf16x16 bA = *(const bf16x16*)bp;
    bf16x16 bB = *(const bf16x16*)(bp + 512);
#pragma unroll
    for (int mt = 0; mt < 3; mt++) {
      bf16x16 a = load_a_lds(m1 + (mt * 16 + ln) * 256 + kt * 32 + g * 8);
      acc2[mt][0] = WMMA_BF16(a, bA, acc2[mt][0]);
      acc2[mt][1] = WMMA_BF16(a, bB, acc2[mt][1]);
    }
  }
  // silu + scatter-add into agg[dst]
#pragma unroll
  for (int mt = 0; mt < 3; mt++)
#pragma unroll
    for (int nj = 0; nj < 2; nj++) {
      const int n = (nt0 + nj) * 16 + ln;
      const float bias = b2[n];
#pragma unroll
      for (int j = 0; j < 8; j++) {
        const int m = mt * 16 + g * 8 + j;
        if (eb + m < E) {
          float x = silu_f(acc2[mt][nj][j] + bias);
          atomicAdd(agg + (size_t)sdst[m] * 256 + n, x);
        }
      }
    }
}

// ---------------------------------------------------------------------------
// Node MLP + residual + LayerNorm. NROWS nodes / block.
// node_in (528 -> pad 544); GEMM1 K=544 (17 kt), GEMM2 K=256.
// ---------------------------------------------------------------------------
__global__ __launch_bounds__(256) void node_mlp_kernel(
    const __bf16* hbf_in, const float* h_in, const float* __restrict__ agg,
    const int* __restrict__ node_role, const int* __restrict__ node_col,
    const float* __restrict__ role_e, const float* __restrict__ col_e,
    const __bf16* __restrict__ w1, const float* __restrict__ b1,
    const __bf16* __restrict__ w2, const float* __restrict__ b2,
    const float* __restrict__ lng, const float* __restrict__ lnb,
    float* h_out, __bf16* hbf_out, int N) {
  __shared__ __align__(16) unsigned char smraw[NROWS * 544 * 2];
  __bf16* sA = (__bf16*)smraw;
  __shared__ int snode[NROWS];
  const int tid = threadIdx.x;
  const int nb = blockIdx.x * NROWS;
  if (tid < NROWS) { int r = nb + tid; snode[tid] = (r < N) ? r : (N - 1); }
  __syncthreads();
  // stage h (bf16 copy)
  for (int idx = tid; idx < NROWS * 32; idx += 256) {
    const int e = idx >> 5, i = idx & 31;
    ((uint4*)(sA + e * 544))[i] = ((const uint4*)(hbf_in + (size_t)snode[e] * 256))[i];
  }
  // stage agg (fp32 -> bf16, float4 granularity)
  for (int idx = tid; idx < NROWS * 64; idx += 256) {
    const int e = idx >> 6, i = idx & 63;
    const float4 f = ((const float4*)(agg + (size_t)snode[e] * 256))[i];
    __bf16* arow = sA + e * 544 + 256 + i * 4;
    arow[0] = (__bf16)f.x; arow[1] = (__bf16)f.y;
    arow[2] = (__bf16)f.z; arow[3] = (__bf16)f.w;
  }
  if (tid < NROWS) {  // tail: role(8) col(8) pad(16)
    const int e = tid, idx = snode[e];
    __bf16* t = sA + e * 544 + 512;
    const float* r0 = role_e + node_role[idx] * 8;
    const float* c0 = col_e + node_col[idx] * 8;
#pragma unroll
    for (int i = 0; i < 8; i++) { t[i] = (__bf16)r0[i]; t[8 + i] = (__bf16)c0[i]; }
#pragma unroll
    for (int i = 0; i < 16; i++) t[16 + i] = (__bf16)0.0f;
  }
  __syncthreads();

  const int lane = tid & 31, wv = tid >> 5;
  const int g = lane >> 4, ln = lane & 15;
  const int nt0 = wv * 2;

  f32x8 acc[3][2] = {};
  for (int kt = 0; kt < 17; ++kt) {
    const __bf16* bp = w1 + ((size_t)((kt * 16 + nt0) * 32 + lane)) * 16;
    __builtin_prefetch(bp + 16 * 32 * 16, 0, 0);  // next kt's B tiles
    bf16x16 bA = *(const bf16x16*)bp;
    bf16x16 bB = *(const bf16x16*)(bp + 512);
#pragma unroll
    for (int mt = 0; mt < 3; mt++) {
      bf16x16 a = load_a_lds(sA + (mt * 16 + ln) * 544 + kt * 32 + g * 8);
      acc[mt][0] = WMMA_BF16(a, bA, acc[mt][0]);
      acc[mt][1] = WMMA_BF16(a, bB, acc[mt][1]);
    }
  }
  __syncthreads();
  __bf16* m1 = sA;
#pragma unroll
  for (int mt = 0; mt < 3; mt++)
#pragma unroll
    for (int nj = 0; nj < 2; nj++) {
      const int n = (nt0 + nj) * 16 + ln;
      const float bias = b1[n];
#pragma unroll
      for (int j = 0; j < 8; j++) {
        float x = acc[mt][nj][j] + bias;
        m1[(mt * 16 + g * 8 + j) * 256 + n] = (__bf16)silu_f(x);
      }
    }
  __syncthreads();

  f32x8 acc2[3][2] = {};
  for (int kt = 0; kt < 8; ++kt) {
    const __bf16* bp = w2 + ((size_t)((kt * 16 + nt0) * 32 + lane)) * 16;
    bf16x16 bA = *(const bf16x16*)bp;
    bf16x16 bB = *(const bf16x16*)(bp + 512);
#pragma unroll
    for (int mt = 0; mt < 3; mt++) {
      bf16x16 a = load_a_lds(m1 + (mt * 16 + ln) * 256 + kt * 32 + g * 8);
      acc2[mt][0] = WMMA_BF16(a, bA, acc2[mt][0]);
      acc2[mt][1] = WMMA_BF16(a, bB, acc2[mt][1]);
    }
  }
  __syncthreads();  // all waves done with m1 before reusing LDS as fp32 x
  float* xs = (float*)smraw;  // NROWS x 256 fp32 = 49152 B <= 52224 B
#pragma unroll
  for (int mt = 0; mt < 3; mt++)
#pragma unroll
    for (int nj = 0; nj < 2; nj++) {
      const int n = (nt0 + nj) * 16 + ln;
      const float bias = b2[n];
#pragma unroll
      for (int j = 0; j < 8; j++) {
        const int m = mt * 16 + g * 8 + j;
        float x = acc2[mt][nj][j] + bias + h_in[(size_t)snode[m] * 256 + n];
        xs[m * 256 + n] = x;
      }
    }
  __syncthreads();
  // LayerNorm: groups of 8 lanes per row, wave32 shfl_xor reduction
  const int q = tid & 7;
  for (int r = tid >> 3; r < NROWS; r += 32) {
    float s1 = 0.f, s2 = 0.f;
    for (int c = q * 32; c < q * 32 + 32; c++) {
      float v = xs[r * 256 + c]; s1 += v; s2 += v * v;
    }
    s1 += __shfl_xor(s1, 1, 32); s1 += __shfl_xor(s1, 2, 32); s1 += __shfl_xor(s1, 4, 32);
    s2 += __shfl_xor(s2, 1, 32); s2 += __shfl_xor(s2, 2, 32); s2 += __shfl_xor(s2, 4, 32);
    const float mu = s1 * (1.0f / 256.0f);
    const float var = s2 * (1.0f / 256.0f) - mu * mu;
    const float rstd = rsqrtf(var + 1e-5f);
    const int gr = nb + r;
    if (gr < N) {
      for (int c = q * 32; c < q * 32 + 32; c++) {
        float v = (xs[r * 256 + c] - mu) * rstd * lng[c] + lnb[c];
        h_out[(size_t)gr * 256 + c] = v;
        hbf_out[(size_t)gr * 256 + c] = (__bf16)v;
      }
    }
  }
}

// ---------------------------------------------------------------------------
// Output projection: out[N,64] = h @ out_w[256,64] + b  (WMMA, OROWS nodes/block)
// ---------------------------------------------------------------------------
__global__ __launch_bounds__(256) void out_proj_kernel(
    const __bf16* __restrict__ hbf, const __bf16* __restrict__ w,
    const float* __restrict__ b, float* __restrict__ out, int N) {
  __shared__ __align__(16) __bf16 sA[OROWS * 256];
  const int tid = threadIdx.x, nb = blockIdx.x * OROWS;
  for (int idx = tid; idx < OROWS * 32; idx += 256) {
    const int e = idx >> 5, i = idx & 31;
    int r = nb + e; int id = (r < N) ? r : (N - 1);
    ((uint4*)(sA + e * 256))[i] = ((const uint4*)(hbf + (size_t)id * 256))[i];
  }
  __syncthreads();
  const int lane = tid & 31, wv = tid >> 5;
  const int g = lane >> 4, ln = lane & 15;
  const int nt = wv & 3, mtb = (wv >> 2) * 2;  // 8 waves = (2x2 m-tiles) x 4 n-tiles
  f32x8 acc[2] = {};
  for (int kt = 0; kt < 8; ++kt) {
    const __bf16* bp = w + ((size_t)((kt * 4 + nt) * 32 + lane)) * 16;
    bf16x16 bb = *(const bf16x16*)bp;
#pragma unroll
    for (int mi = 0; mi < 2; mi++) {
      bf16x16 a = load_a_lds(sA + ((mtb + mi) * 16 + ln) * 256 + kt * 32 + g * 8);
      acc[mi] = WMMA_BF16(a, bb, acc[mi]);
    }
  }
  const int n = nt * 16 + ln;
  const float bias = b[n];
#pragma unroll
  for (int mi = 0; mi < 2; mi++)
#pragma unroll
    for (int j = 0; j < 8; j++) {
      const int r = nb + (mtb + mi) * 16 + g * 8 + j;
      if (r < N) out[(size_t)r * 64 + n] = acc[mi][j] + bias;
    }
}

// ---------------------------------------------------------------------------
extern "C" void kernel_launch(void* const* d_in, const int* in_sizes, int n_in,
                              void* d_out, int out_size, void* d_ws, size_t ws_size,
                              hipStream_t stream) {
  (void)n_in; (void)out_size; (void)ws_size;
  const float* scalars    = (const float*)d_in[0];
  const int*   edge_index = (const int*)d_in[1];
  const int*   edge_rel   = (const int*)d_in[2];
  const int*   node_col   = (const int*)d_in[3];
  const int*   node_role  = (const int*)d_in[4];
  const float* blk_role   = (const float*)d_in[5];
  const float* blk_col    = (const float*)d_in[6];
  const float* in_w       = (const float*)d_in[7];
  const float* in_b       = (const float*)d_in[8];
  const float* rel_embs   = (const float*)d_in[9];
  const float* role_embs  = (const float*)d_in[10];
  const float* col_embs   = (const float*)d_in[11];
  const float* e1w        = (const float*)d_in[12];
  const float* e1b        = (const float*)d_in[13];
  const float* e2w        = (const float*)d_in[14];
  const float* e2b        = (const float*)d_in[15];
  const float* n1w        = (const float*)d_in[16];
  const float* n1b        = (const float*)d_in[17];
  const float* n2w        = (const float*)d_in[18];
  const float* n2b        = (const float*)d_in[19];
  const float* lng        = (const float*)d_in[20];
  const float* lnb        = (const float*)d_in[21];
  const float* outw       = (const float*)d_in[22];
  const float* outb       = (const float*)d_in[23];

  const int E = in_sizes[2];  // edge_relation length
  const int N = in_sizes[3];  // node_color_rep length
  const int L = 3, H = 256;

  // workspace layout
  char* ws = (char*)d_ws; size_t off = 0;
  auto alloc = [&](size_t bytes) -> void* {
    void* p = ws + off; off = (off + bytes + 255) & ~(size_t)255; return p;
  };
  float*  h   = (float*)alloc((size_t)N * H * 4);
  __bf16* hbf = (__bf16*)alloc((size_t)N * H * 2);
  float*  agg = (float*)alloc((size_t)N * H * 4);
  const int E1T = 18 * 16 * 512, E2T = 8 * 16 * 512;  // halves per layer
  const int N1T = 17 * 16 * 512, N2T = 8 * 16 * 512;
  const int OWT = 8 * 4 * 512;
  __bf16* e1ws = (__bf16*)alloc((size_t)L * E1T * 2);
  __bf16* e2ws = (__bf16*)alloc((size_t)L * E2T * 2);
  __bf16* n1ws = (__bf16*)alloc((size_t)L * N1T * 2);
  __bf16* n2ws = (__bf16*)alloc((size_t)L * N2T * 2);
  __bf16* owws = (__bf16*)alloc((size_t)OWT * 2);

  auto swz = [&](const float* src, __bf16* dst, int Kact, int NT, int ncols, int total) {
    swizzle_kernel<<<(total + 255) / 256, 256, 0, stream>>>(src, dst, Kact, NT, ncols, total);
  };
  for (int l = 0; l < L; ++l) {
    swz(e1w + (size_t)l * 560 * 256, e1ws + (size_t)l * E1T, 560, 16, 256, E1T);
    swz(e2w + (size_t)l * 256 * 256, e2ws + (size_t)l * E2T, 256, 16, 256, E2T);
    swz(n1w + (size_t)l * 528 * 256, n1ws + (size_t)l * N1T, 528, 16, 256, N1T);
    swz(n2w + (size_t)l * 256 * 256, n2ws + (size_t)l * N2T, 256, 16, 256, N2T);
  }
  swz(outw, owws, 256, 4, 64, OWT);

  input_proj_kernel<<<N, 256, 0, stream>>>(scalars, node_col, node_role, blk_col,
                                           blk_role, in_w, in_b, h, hbf, N);

  const int edge_blocks = (E + EROWS - 1) / EROWS;
  const int node_blocks = (N + NROWS - 1) / NROWS;
  const int out_blocks  = (N + OROWS - 1) / OROWS;
  for (int l = 0; l < L; ++l) {
    zero_kernel<<<2048, 256, 0, stream>>>(agg, (size_t)N * H);
    edge_mlp_kernel<<<edge_blocks, 256, 0, stream>>>(
        hbf, edge_index, edge_rel, node_role, node_col,
        rel_embs + (size_t)l * 8 * 16, role_embs + (size_t)l * 8 * 8,
        col_embs + (size_t)l * 3 * 8,
        e1ws + (size_t)l * E1T, e1b + (size_t)l * 256,
        e2ws + (size_t)l * E2T, e2b + (size_t)l * 256, agg, E);
    node_mlp_kernel<<<node_blocks, 256, 0, stream>>>(
        hbf, h, agg, node_role, node_col,
        role_embs + (size_t)l * 8 * 8, col_embs + (size_t)l * 3 * 8,
        n1ws + (size_t)l * N1T, n1b + (size_t)l * 256,
        n2ws + (size_t)l * N2T, n2b + (size_t)l * 256,
        lng + (size_t)l * 256, lnb + (size_t)l * 256, h, hbf, N);
  }
  out_proj_kernel<<<out_blocks, 256, 0, stream>>>(hbf, owws, outb, (float*)d_out, N);
}
